// CrystalGCNEncoder_16260746183179
// MI455X (gfx1250) — compile-verified, matrix operand order checked by reading, not run on hardware
//
#include <hip/hip_runtime.h>
#include <hip/hip_bf16.h>

typedef __attribute__((ext_vector_type(16))) __bf16 v16bf;
typedef __attribute__((ext_vector_type(8)))  float  v8f;

// Problem constants (match reference)
constexpr int N_NODES = 100000;
constexpr int N_EDGES = 800000;
constexpr int N_GRAPH = 64;
constexpr int HEADS   = 4;
constexpr int F_IN    = 92;
constexpr int FINAL   = 512;   // H * D3
constexpr int HID     = 64;
constexpr int LAT     = 128;
constexpr float NEG_SLOPE = 0.2f;

// ---------- ordered-float key helpers (for atomic segment max) ----------
__device__ __forceinline__ unsigned f2key(float f) {
  unsigned u = __float_as_uint(f);
  return (u & 0x80000000u) ? ~u : (u | 0x80000000u);
}
__device__ __forceinline__ float key2f(unsigned k) {
  return (k & 0x80000000u) ? __uint_as_float(k ^ 0x80000000u)
                           : __uint_as_float(~k);
}
constexpr unsigned NEG_INF_KEY = 0x007FFFFFu;  // f2key(-inf)

// ---------- generic fills ----------
__global__ void fill_u32(unsigned* p, unsigned v, long long n) {
  long long i = blockIdx.x * (long long)blockDim.x + threadIdx.x;
  if (i < n) p[i] = v;
}
__global__ void fill_f32(float* p, float v, long long n) {
  long long i = blockIdx.x * (long long)blockDim.x + threadIdx.x;
  if (i < n) p[i] = v;
}

// ---------- activations fp32 -> bf16 with K padding (row major) ----------
__global__ void act_to_bf16(const float* __restrict__ in, __bf16* __restrict__ out,
                            int rows, int K, int Kp) {
  long long i = blockIdx.x * (long long)blockDim.x + threadIdx.x;
  long long total = (long long)rows * Kp;
  if (i >= total) return;
  int r = (int)(i / Kp), c = (int)(i % Kp);
  out[i] = (c < K) ? (__bf16)in[(size_t)r * K + c] : (__bf16)0.0f;
}

// ---------- weights fp32 -> WMMA B-fragment-packed bf16 ----------
// Layout: Bp[(((kt*M + col)*2 + half)*16) + e] holds B[k][col] with
//   e<8 : k = kt*32 + half*8 + e
//   e>=8: k = kt*32 + half*8 + 8 + e   (i.e. +16 + (e-8))
// so each lane's 16 fragment values are contiguous (two b128 loads).
__global__ void w_pack_frag(const float* __restrict__ W, __bf16* __restrict__ Bp,
                            int Kin, int Kp, int M) {
  long long i = blockIdx.x * (long long)blockDim.x + threadIdx.x;
  long long total = (long long)Kp * M;
  if (i >= total) return;
  int e = (int)(i & 15);
  long long r = i >> 4;
  int half = (int)(r & 1); r >>= 1;
  int col = (int)(r % M);
  int kt  = (int)(r / M);
  int k = kt * 32 + half * 8 + ((e < 8) ? e : 8 + e);
  Bp[i] = (k < Kin) ? (__bf16)W[(size_t)k * M + col] : (__bf16)0.0f;
}

// ---------- WMMA bf16 GEMM: C[N x M] = A[N x Kp] * B[Kp x M] ----------
// One wave computes a 16x64 strip: 4 column tiles sharing one A fragment.
// Rows (N) and cols (M) are exact multiples of 16; M multiple of 64.
__global__ void __launch_bounds__(32)
gemm_bf16_wmma4(const __bf16* __restrict__ A, const __bf16* __restrict__ Bp,
                float* __restrict__ C, int Kp, int M) {
  const int tn0  = blockIdx.x * 4;    // first column tile of this wave
  const int tm   = blockIdx.y;        // row tile
  const int lane = threadIdx.x;
  const int half = lane >> 4;         // 0: lanes 0-15, 1: lanes 16-31
  const int l15  = lane & 15;
  const int rowA = tm * 16 + l15;

  v8f acc[4] = {};
  const int nkt = Kp >> 5;
  for (int kt = 0; kt < nkt; ++kt) {
    const int k0 = kt * 32;
    v16bf a;
    const __bf16* ap = A + (size_t)rowA * Kp + k0 + half * 8;
#pragma unroll
    for (int e = 0; e < 8; ++e) {
      a[e]     = ap[e];        // K = k0 + half*8 + e
      a[e + 8] = ap[16 + e];   // K = k0 + 16 + half*8 + e
    }
#pragma unroll
    for (int t = 0; t < 4; ++t) {
      const int col = (tn0 + t) * 16 + l15;
      const __bf16* bp = Bp + ((((size_t)kt * M + col) * 2 + half) << 4);
      v16bf b;
#pragma unroll
      for (int e = 0; e < 16; ++e) b[e] = bp[e];
      acc[t] = __builtin_amdgcn_wmma_f32_16x16x32_bf16(
          false, a, false, b, (short)0, acc[t], false, false);
    }
  }
  // C/D layout: VGPR r -> row (r + half*8), col = lane&15
#pragma unroll
  for (int t = 0; t < 4; ++t) {
#pragma unroll
    for (int r = 0; r < 8; ++r) {
      int rowC = tm * 16 + r + half * 8;
      C[(size_t)rowC * M + (tn0 + t) * 16 + l15] = acc[t][r];
    }
  }
}

// ---------- el / er : per-(node,head) dot with attention vectors ----------
__global__ void compute_elr(const float* __restrict__ Wh,
                            const float* __restrict__ al, const float* __restrict__ ar,
                            float* __restrict__ el, float* __restrict__ er,
                            int D, int HD) {
  int i = blockIdx.x * blockDim.x + threadIdx.x;   // n*HEADS + h
  if (i >= N_NODES * HEADS) return;
  int n = i / HEADS, hh = i % HEADS;
  const float* row = Wh + (size_t)n * HD + hh * D;
  const float* va  = al + hh * D;
  const float* vb  = ar + hh * D;
  float sl = 0.f, sr = 0.f;
  for (int d = 0; d < D; ++d) { float v = row[d]; sl += v * va[d]; sr += v * vb[d]; }
  el[i] = sl; er[i] = sr;
}

// ---------- edge softmax passes ----------
__global__ void edge_max(const int* __restrict__ src, const int* __restrict__ dst,
                         const float* __restrict__ el, const float* __restrict__ er,
                         float* __restrict__ ebuf, unsigned* __restrict__ mkey) {
  long long i = blockIdx.x * (long long)blockDim.x + threadIdx.x;
  if (i >= (long long)N_EDGES * HEADS) return;
  int e = (int)(i / HEADS), hh = (int)(i % HEADS);
  float v = el[src[e] * HEADS + hh] + er[dst[e] * HEADS + hh];
  v = (v > 0.f) ? v : NEG_SLOPE * v;               // leaky_relu
  ebuf[i] = v;
  atomicMax(&mkey[dst[e] * HEADS + hh], f2key(v));
}
__global__ void decode_max(unsigned* __restrict__ mkey, long long n) {
  long long i = blockIdx.x * (long long)blockDim.x + threadIdx.x;
  if (i >= n) return;
  unsigned k = mkey[i];
  float m = (k == NEG_INF_KEY) ? 0.0f : key2f(k);  // zero-in-degree guard
  mkey[i] = __float_as_uint(m);                    // reuse buffer as float
}
__global__ void edge_exp(const int* __restrict__ dst,
                         const float* __restrict__ mval,
                         float* __restrict__ ebuf, float* __restrict__ ssum) {
  long long i = blockIdx.x * (long long)blockDim.x + threadIdx.x;
  if (i >= (long long)N_EDGES * HEADS) return;
  int e = (int)(i / HEADS), hh = (int)(i % HEADS);
  float ex = __expf(ebuf[i] - mval[dst[e] * HEADS + hh]);
  ebuf[i] = ex;
  atomicAdd(&ssum[dst[e] * HEADS + hh], ex);
}

// ---------- bias init + scatter-aggregate + ELU ----------
__global__ void init_bias(float* __restrict__ out, const float* __restrict__ b,
                          int HD) {
  long long i = blockIdx.x * (long long)blockDim.x + threadIdx.x;
  if (i >= (long long)N_NODES * HD) return;
  out[i] = b[(int)(i % HD)];
}
__global__ void edge_agg(const int* __restrict__ src, const int* __restrict__ dst,
                         const float* __restrict__ Wh, const float* __restrict__ ebuf,
                         const float* __restrict__ ssum, float* __restrict__ out,
                         int D, int HD) {
  int e = blockIdx.x;
  int s = src[e], d = dst[e];
  for (int j = threadIdx.x; j < HD; j += blockDim.x) {
    int hh = j / D;
    float a = ebuf[(size_t)e * HEADS + hh] /
              fmaxf(ssum[(size_t)d * HEADS + hh], 1e-9f);
    atomicAdd(&out[(size_t)d * HD + j], a * Wh[(size_t)s * HD + j]);
  }
}
__global__ void elu_inplace(float* __restrict__ p, long long n) {
  long long i = blockIdx.x * (long long)blockDim.x + threadIdx.x;
  if (i >= n) return;
  float x = p[i];
  p[i] = (x > 0.f) ? x : (__expf(x) - 1.0f);
}

// ---------- gate finish: relu(hid + bg1) . Wg2 + bg2 ----------
__global__ void gate_finish(const float* __restrict__ hid,
                            const float* __restrict__ bg1,
                            const float* __restrict__ Wg2,
                            const float* __restrict__ bg2,
                            float* __restrict__ gate) {
  int n = blockIdx.x * blockDim.x + threadIdx.x;
  if (n >= N_NODES) return;
  const float* row = hid + (size_t)n * HID;
  float acc = 0.f;
#pragma unroll 8
  for (int j = 0; j < HID; ++j)
    acc += fmaxf(row[j] + bg1[j], 0.f) * Wg2[j];
  gate[n] = acc + bg2[0];
}

// ---------- graph-level softmax pooling ----------
__global__ void node_gmax(const int* __restrict__ gid, const float* __restrict__ gate,
                          unsigned* __restrict__ gmk) {
  int n = blockIdx.x * blockDim.x + threadIdx.x;
  if (n >= N_NODES) return;
  atomicMax(&gmk[gid[n]], f2key(gate[n]));
}
__global__ void node_gexp(const int* __restrict__ gid, const float* __restrict__ gate,
                          const float* __restrict__ gm, float* __restrict__ exn,
                          float* __restrict__ gss) {
  int n = blockIdx.x * blockDim.x + threadIdx.x;
  if (n >= N_NODES) return;
  float ex = __expf(gate[n] - gm[gid[n]]);
  exn[n] = ex;
  atomicAdd(&gss[gid[n]], ex);
}
__global__ void pool_sum(const int* __restrict__ gid, const float* __restrict__ h,
                         const float* __restrict__ exn, const float* __restrict__ gss,
                         float* __restrict__ pooled) {
  long long i = blockIdx.x * (long long)blockDim.x + threadIdx.x;
  if (i >= (long long)N_NODES * FINAL) return;
  int n = (int)(i / FINAL), f = (int)(i % FINAL);
  int g = gid[n];
  float a = exn[n] / fmaxf(gss[g], 1e-9f);
  atomicAdd(&pooled[(size_t)g * FINAL + f], a * h[i]);
}

// ---------- final heads: mu & logvar ----------
__global__ void final_heads(const float* __restrict__ pooled,
                            const float* __restrict__ Wmu, const float* __restrict__ bmu,
                            const float* __restrict__ Wlv, const float* __restrict__ blv,
                            float* __restrict__ out) {
  int i = blockIdx.x * blockDim.x + threadIdx.x;      // over 2*G*LAT
  if (i >= 2 * N_GRAPH * LAT) return;
  int which = i / (N_GRAPH * LAT);
  int rem = i % (N_GRAPH * LAT);
  int g = rem / LAT, j = rem % LAT;
  const float* W = which ? Wlv : Wmu;
  const float* b = which ? blv : bmu;
  const float* row = pooled + (size_t)g * FINAL;
  float acc = 0.f;
  for (int k = 0; k < FINAL; ++k) acc += row[k] * W[(size_t)k * LAT + j];
  out[i] = acc + b[j];
}

// ===================== host-side driver =====================
static inline size_t alignup(size_t x) { return (x + 255) & ~size_t(255); }
static inline long long cdivll(long long a, long long b) { return (a + b - 1) / b; }

extern "C" void kernel_launch(void* const* d_in, const int* in_sizes, int n_in,
                              void* d_out, int out_size, void* d_ws, size_t ws_size,
                              hipStream_t stream) {
  (void)in_sizes; (void)n_in; (void)out_size; (void)ws_size;

  const float* node_feat = (const float*)d_in[0];
  const int*   src       = (const int*)d_in[1];
  const int*   dst       = (const int*)d_in[2];
  const int*   gid       = (const int*)d_in[3];
  struct LayerP { const float *W, *al, *ar, *b; };
  LayerP L[3] = {
    { (const float*)d_in[4],  (const float*)d_in[5],  (const float*)d_in[6],  (const float*)d_in[7]  },
    { (const float*)d_in[8],  (const float*)d_in[9],  (const float*)d_in[10], (const float*)d_in[11] },
    { (const float*)d_in[12], (const float*)d_in[13], (const float*)d_in[14], (const float*)d_in[15] },
  };
  const float* Wg1 = (const float*)d_in[16];
  const float* bg1 = (const float*)d_in[17];
  const float* Wg2 = (const float*)d_in[18];
  const float* bg2 = (const float*)d_in[19];
  const float* Wmu = (const float*)d_in[20];
  const float* bmu = (const float*)d_in[21];
  const float* Wlv = (const float*)d_in[22];
  const float* blv = (const float*)d_in[23];
  float* out = (float*)d_out;

  // workspace carve-up
  char* ws = (char*)d_ws;
  size_t off = 0;
  float*    bufA  = (float*)(ws + off);   off = alignup(off + (size_t)N_NODES * 512 * 4);
  float*    bufB  = (float*)(ws + off);   off = alignup(off + (size_t)N_NODES * 512 * 4);
  __bf16*   hbf   = (__bf16*)(ws + off);  off = alignup(off + (size_t)N_NODES * 512 * 2);
  __bf16*   wpack = (__bf16*)(ws + off);  off = alignup(off + (size_t)512 * 512 * 2);
  float*    el    = (float*)(ws + off);   off = alignup(off + (size_t)N_NODES * HEADS * 4);
  float*    er    = (float*)(ws + off);   off = alignup(off + (size_t)N_NODES * HEADS * 4);
  unsigned* mkey  = (unsigned*)(ws + off);off = alignup(off + (size_t)N_NODES * HEADS * 4);
  float*    ssum  = (float*)(ws + off);   off = alignup(off + (size_t)N_NODES * HEADS * 4);
  float*    ebuf  = (float*)(ws + off);   off = alignup(off + (size_t)N_EDGES * HEADS * 4);
  float*    gate  = (float*)(ws + off);   off = alignup(off + (size_t)N_NODES * 4);
  float*    exn   = (float*)(ws + off);   off = alignup(off + (size_t)N_NODES * 4);
  unsigned* gmk   = (unsigned*)(ws + off);off = alignup(off + (size_t)N_GRAPH * 4);
  float*    gss   = (float*)(ws + off);   off = alignup(off + (size_t)N_GRAPH * 4);
  float*    pooled= (float*)(ws + off);   off = alignup(off + (size_t)N_GRAPH * FINAL * 4);

  const int BT = 256;
  struct Cfg { int Kin, Kp, M, D; };
  Cfg cfg[3] = { {F_IN, 96, 128, 32}, {128, 128, 256, 64}, {256, 256, 512, 128} };

  for (int l = 0; l < 3; ++l) {
    const Cfg c = cfg[l];
    const float* hin = (l == 0) ? node_feat : bufA;

    // bf16 conversions (zero-padded K); weights packed to fragment layout
    long long na = (long long)N_NODES * c.Kp;
    act_to_bf16<<<cdivll(na, BT), BT, 0, stream>>>(hin, hbf, N_NODES, c.Kin, c.Kp);
    long long nw = (long long)c.Kp * c.M;
    w_pack_frag<<<cdivll(nw, BT), BT, 0, stream>>>(L[l].W, wpack, c.Kin, c.Kp, c.M);

    // Wh = h @ W via WMMA bf16 (one wave per 16x64 strip)
    dim3 gg(c.M / 64, N_NODES / 16);
    gemm_bf16_wmma4<<<gg, 32, 0, stream>>>(hbf, wpack, bufB, c.Kp, c.M);

    // attention logits
    compute_elr<<<cdivll((long long)N_NODES * HEADS, BT), BT, 0, stream>>>(
        bufB, L[l].al, L[l].ar, el, er, c.D, c.M);

    // edge softmax
    long long nh = (long long)N_NODES * HEADS;
    long long eh = (long long)N_EDGES * HEADS;
    fill_u32<<<cdivll(nh, BT), BT, 0, stream>>>(mkey, NEG_INF_KEY, nh);
    fill_f32<<<cdivll(nh, BT), BT, 0, stream>>>(ssum, 0.0f, nh);
    edge_max<<<cdivll(eh, BT), BT, 0, stream>>>(src, dst, el, er, ebuf, mkey);
    decode_max<<<cdivll(nh, BT), BT, 0, stream>>>(mkey, nh);
    edge_exp<<<cdivll(eh, BT), BT, 0, stream>>>(dst, (const float*)mkey, ebuf, ssum);

    // aggregate: out = bias + sum_e a * Wh[src]
    long long nm = (long long)N_NODES * c.M;
    init_bias<<<cdivll(nm, BT), BT, 0, stream>>>(bufA, L[l].b, c.M);
    edge_agg<<<N_EDGES, 128, 0, stream>>>(src, dst, bufB, ebuf, ssum, bufA, c.D, c.M);
    elu_inplace<<<cdivll(nm, BT), BT, 0, stream>>>(bufA, nm);
  }

  // gate MLP: hid = h @ Wg1 via WMMA, then relu/dot finish
  act_to_bf16<<<cdivll((long long)N_NODES * FINAL, BT), BT, 0, stream>>>(
      bufA, hbf, N_NODES, FINAL, FINAL);
  w_pack_frag<<<cdivll((long long)FINAL * HID, BT), BT, 0, stream>>>(
      Wg1, wpack, FINAL, FINAL, HID);
  dim3 gg_gate(HID / 64, N_NODES / 16);
  gemm_bf16_wmma4<<<gg_gate, 32, 0, stream>>>(hbf, wpack, bufB, FINAL, HID);
  gate_finish<<<cdivll(N_NODES, BT), BT, 0, stream>>>(bufB, bg1, Wg2, bg2, gate);

  // attention pooling over graphs
  fill_u32<<<1, N_GRAPH, 0, stream>>>(gmk, NEG_INF_KEY, N_GRAPH);
  fill_f32<<<1, N_GRAPH, 0, stream>>>(gss, 0.0f, N_GRAPH);
  fill_f32<<<cdivll((long long)N_GRAPH * FINAL, BT), BT, 0, stream>>>(
      pooled, 0.0f, (long long)N_GRAPH * FINAL);
  node_gmax<<<cdivll(N_NODES, BT), BT, 0, stream>>>(gid, gate, gmk);
  decode_max<<<1, N_GRAPH, 0, stream>>>(gmk, N_GRAPH);
  node_gexp<<<cdivll(N_NODES, BT), BT, 0, stream>>>(gid, gate, (const float*)gmk, exn, gss);
  pool_sum<<<cdivll((long long)N_NODES * FINAL, BT), BT, 0, stream>>>(
      gid, bufA, exn, gss, pooled);

  // mu / logvar heads -> d_out (mu first, then logvar)
  final_heads<<<cdivll(2 * N_GRAPH * LAT, BT), BT, 0, stream>>>(
      pooled, Wmu, bmu, Wlv, blv, out);
}
